// XNORConv2d_7370163880459
// MI455X (gfx1250) — 8-bit matrix operand order repaired and checked by reading, compiled, not run on hardware
//
#include <hip/hip_runtime.h>
#include <stdint.h>

typedef int v8i __attribute__((ext_vector_type(8)));

#define N_  32
#define C_  64
#define O_  64
#define H_  128
#define W_  128
#define HP  130
#define WP  130

static const size_t XBP_BYTES  = (size_t)N_ * HP * WP * C_;
static const size_t WSGN_BYTES = (size_t)9 * O_ * C_;

__global__ void xnor_prep_weights(const float* __restrict__ fp,
                                  int8_t* __restrict__ wsgn,
                                  float* __restrict__ meanv) {
    int o = threadIdx.x;
    if (o >= O_) return;
    const float* wo = fp + (size_t)o * (C_ * 9);
    float tm[9];
    #pragma unroll
    for (int t = 0; t < 9; ++t) {
        float s = 0.f;
        for (int c = 0; c < C_; ++c) s += wo[c * 9 + t];
        tm[t] = s * (1.0f / C_);
    }
    float asum = 0.f;
    for (int c = 0; c < C_; ++c) {
        #pragma unroll
        for (int t = 0; t < 9; ++t) {
            float w = wo[c * 9 + t] - tm[t];
            w = fminf(1.0f, fmaxf(-1.0f, w));
            asum += fabsf(w);
            wsgn[(size_t)t * (O_ * C_) + o * C_ + c] =
                (int8_t)((w > 0.f) - (w < 0.f));
        }
    }
    meanv[o] = asum * (1.0f / (C_ * 9));
}

__global__ void __launch_bounds__(256) xnor_binarize(const float* __restrict__ x,
                                                     int8_t* __restrict__ xbp) {
    const int b  = blockIdx.x;
    const int wt = b & 1;
    const int h  = (b >> 1) & (H_ - 1);
    const int n  = b >> 8;
    const int w0 = wt * 64;
    const int t  = threadIdx.x;

    __shared__ int8_t tile[64 * 72];

    const float* xb = x + (((size_t)n * C_) * H_ + h) * W_ + w0;
    #pragma unroll
    for (int i = 0; i < 16; ++i) {
        int idx = i * 256 + t;
        int c = idx >> 6, w = idx & 63;
        float v = xb[(size_t)c * (H_ * W_) + w];
        tile[w * 72 + c] = (int8_t)((v > 0.f) - (v < 0.f));
    }
    __syncthreads();

    int8_t* op = xbp + (((size_t)n * HP + (h + 1)) * WP + (w0 + 1)) * C_;
    #pragma unroll
    for (int i = 0; i < 4; ++i) {
        int idx = i * 256 + t;
        int cq = idx & 15, w = idx >> 4;
        *(int*)(op + (size_t)w * C_ + cq * 4) =
            *(const int*)(tile + w * 72 + cq * 4);
    }
}

__global__ void __launch_bounds__(256) xnor_conv_wmma(
        const int8_t* __restrict__ xbp, const int8_t* __restrict__ wsgn,
        const float* __restrict__ meanv, const float* __restrict__ bias,
        float* __restrict__ out) {
    const int lane = threadIdx.x & 31;
    const int hi   = lane >> 4;
    const int m    = lane & 15;
    const int wave = blockIdx.x * 8 + (threadIdx.x >> 5);
    const int wt = wave & 3;
    const int h  = (wave >> 2) & (H_ - 1);
    const int n  = wave >> 9;
    const int w0 = wt * 32;

    v8i acc[8] = {v8i{}, v8i{}, v8i{}, v8i{}, v8i{}, v8i{}, v8i{}, v8i{}};

    const int8_t* abase = xbp +
        (((size_t)n * HP + h) * WP + (w0 + m)) * C_ + hi * 8;
    const int8_t* bbase = wsgn + m * C_ + hi * 8;

    #pragma unroll
    for (int kh = 0; kh < 3; ++kh) {
        #pragma unroll
        for (int kw = 0; kw < 3; ++kw) {
            const int tap  = kh * 3 + kw;
            const int aoff = (kh * WP + kw) * C_;
            const int8_t* ar0 = abase + aoff;
            const int8_t* ar1 = abase + aoff + 16*C_;
            int2 a00 = *(const int2*)(ar0 +  0);
            int2 a01 = *(const int2*)(ar0 + 16);
            int2 a02 = *(const int2*)(ar0 + 32);
            int2 a03 = *(const int2*)(ar0 + 48);
            v8i aLo = {a00.x, a00.y, a01.x, a01.y, a02.x, a02.y, a03.x, a03.y};
            int2 a10 = *(const int2*)(ar1 +  0);
            int2 a11 = *(const int2*)(ar1 + 16);
            int2 a12 = *(const int2*)(ar1 + 32);
            int2 a13 = *(const int2*)(ar1 + 48);
            v8i aHi = {a10.x, a10.y, a11.x, a11.y, a12.x, a12.y, a13.x, a13.y};

            #pragma unroll
            for (int ot = 0; ot < 4; ++ot) {
                const int8_t* bb = bbase + (size_t)tap * (O_ * C_) + ot * 16 * C_;
                int2 b0 = *(const int2*)(bb +  0);
                int2 b1 = *(const int2*)(bb + 16);
                int2 b2 = *(const int2*)(bb + 32);
                int2 b3 = *(const int2*)(bb + 48);
                v8i bf = {b0.x, b0.y, b1.x, b1.y, b2.x, b2.y, b3.x, b3.y};
                acc[ot]     = __builtin_amdgcn_wmma_i32_16x16x64_iu8(
                    true, aLo, true, bf, acc[ot],     false, false);
                acc[ot + 4] = __builtin_amdgcn_wmma_i32_16x16x64_iu8(
                    true, aHi, true, bf, acc[ot + 4], false, false);
            }
        }
    }

    float* obase = out + (((size_t)n * O_) * H_ + h) * W_ + w0 + hi * 8;
    #pragma unroll
    for (int ot = 0; ot < 4; ++ot) {
        const int o = ot * 16 + m;
        const float sc = meanv[o];
        const float bs = bias[o];
        float* op = obase + (size_t)o * (H_ * W_);
        #pragma unroll
        for (int half = 0; half < 2; ++half) {
            const v8i a = acc[ot + 4 * half];
            float4 lo = { (float)a[0] * sc + bs, (float)a[1] * sc + bs,
                          (float)a[2] * sc + bs, (float)a[3] * sc + bs };
            float4 hi4 = { (float)a[4] * sc + bs, (float)a[5] * sc + bs,
                           (float)a[6] * sc + bs, (float)a[7] * sc + bs };
            *(float4*)(op + half * 16)     = lo;
            *(float4*)(op + half * 16 + 4) = hi4;
        }
    }
}

extern "C" void kernel_launch(void* const* d_in, const int* in_sizes, int n_in,
                              void* d_out, int out_size, void* d_ws, size_t ws_size,
                              hipStream_t stream) {
    const float* x    = (const float*)d_in[0];
    const float* fp   = (const float*)d_in[1];
    const float* bias = (const float*)d_in[2];
    float* out = (float*)d_out;

    int8_t* xbp  = (int8_t*)d_ws;
    int8_t* wsgn = xbp + XBP_BYTES;
    float*  mv   = (float*)(wsgn + WSGN_BYTES);

    hipMemsetAsync(xbp, 0, XBP_BYTES, stream);

    xnor_prep_weights<<<1, 64, 0, stream>>>(fp, wsgn, mv);
    xnor_binarize<<<N_ * H_ * 2, 256, 0, stream>>>(x, xbp);
    xnor_conv_wmma<<<2048, 256, 0, stream>>>(xbp, wsgn, mv, bias, out);
}
